// Attention_6837587935825
// MI455X (gfx1250) — matrix operand order repaired and checked by reading, not compiled, not run on hardware
//
#include <hip/hip_runtime.h>

#define B_ 128
#define T_ 2048
#define H_ 512
#define M_ (B_*T_)          // 262144 rows in the flattened GEMM
#define ROWS 128            // M-rows per block (8 waves x 16)
#define PAD  16
#define LDSW (H_ + PAD)     // 528 bf16 per LDS row -> 1056B, 16B-aligned rows
#define NB   4              // N-chunks (16 cols each) per W-staging phase
#define GROUPS (H_/(16*NB)) // 8 staging phases of 64 N-columns

typedef __bf16 bf16;
typedef __attribute__((ext_vector_type(4)))  bf16  v4bf;
typedef __attribute__((ext_vector_type(8)))  bf16  v8bf;
typedef __attribute__((ext_vector_type(16))) bf16  v16bf;
typedef __attribute__((ext_vector_type(8)))  float v8f;

#if defined(__HIP_DEVICE_COMPILE__) && __has_builtin(__builtin_amdgcn_tensor_load_to_lds)
#define USE_TDM 1
typedef unsigned int u32x4 __attribute__((ext_vector_type(4)));
typedef int          i32x8 __attribute__((ext_vector_type(8)));
typedef int          i32x4 __attribute__((ext_vector_type(4)));
#endif

#if defined(__HIP_DEVICE_COMPILE__) && __has_builtin(__builtin_amdgcn_tanhf)
#define TANHF(x) __builtin_amdgcn_tanhf(x)   // v_tanh_f32 (TRANS, co-executes w/ XDL)
#else
#define TANHF(x) tanhf(x)
#endif

// ---------------- prep: Wsum = Wa+Ua (bf16), bsum = ba+bu, zero context ----
__global__ void prep_kernel(const float* __restrict__ Wa_w,
                            const float* __restrict__ Ua_w,
                            const float* __restrict__ Wa_b,
                            const float* __restrict__ Ua_b,
                            bf16*  __restrict__ Wsum,
                            float* __restrict__ bsum,
                            float* __restrict__ ctx) {
  int i = blockIdx.x * blockDim.x + threadIdx.x;
  if (i < H_*H_) Wsum[i] = (bf16)(Wa_w[i] + Ua_w[i]);
  if (i < H_)    bsum[i] = Wa_b[i] + Ua_b[i];
  if (i < B_*H_) ctx[i]  = 0.0f;
}

#ifdef USE_TDM
// One 2-D TDM descriptor: 64 rows x 512 bf16, dense in global, 528-elem padded
// rows in LDS (pad 8 DWORDs after every 256 DWORDs).
__device__ __forceinline__ void tdm_load_w(const bf16* gsrc, unsigned lds_base) {
  unsigned long long ga = (unsigned long long)(const void*)gsrc;
  u32x4 g0;
  g0.x = 1u;                                          // count=1, load, no gather
  g0.y = lds_base;                                    // lds_addr (bytes)
  g0.z = (unsigned)(ga & 0xffffffffu);                // global_addr[31:0]
  g0.w = (unsigned)((ga >> 32) & 0x01ffffffu)         // global_addr[56:32]
       | (2u << 30);                                  // type = 2 (image)
  i32x8 g1;
  g1[0] = (1 << 16)        // data_size = 2 bytes (bf16)
        | (1 << 20)        // pad_enable
        | (7 << 22)        // pad_interval: every 256 DWORDs (=1024B row)
        | (7 << 25);       // pad_amount: 8 DWORDs (=32B -> 528-elem stride)
  g1[1] = (int)(512u << 16);   // tensor_dim0 = 512
  g1[2] = (int)(64u  << 16);   // tensor_dim1 = 64
  g1[3] = (int)(512u << 16);   // tile_dim0 = 512
  g1[4] = 64;                  // tile_dim1 = 64 ; tile_dim2 = 0
  g1[5] = 512;                 // tensor_dim0_stride = 512 elements
  g1[6] = 0;
  g1[7] = 0;
  i32x4 z4 = {0,0,0,0};
#if __clang_major__ >= 23
  i32x8 z8 = {0,0,0,0,0,0,0,0};
  __builtin_amdgcn_tensor_load_to_lds(g0, g1, z4, z4, z8, 0);
#else
  __builtin_amdgcn_tensor_load_to_lds(g0, g1, z4, z4, 0);
#endif
}
#endif

// ---------------- fused GEMM + tanh + Va-dot -> score[M] --------------------
__global__ __launch_bounds__(256, 1)
void score_kernel(const float* __restrict__ X,     // [M_, H_] fp32
                  const bf16*  __restrict__ Wsum,  // [H_, H_] bf16 row-major (o,h)
                  const float* __restrict__ bsum,  // [H_]
                  const float* __restrict__ VaW,   // [H_]
                  const float* __restrict__ VaB,   // [1]
                  float* __restrict__ score)       // [M_]
{
  __shared__ bf16 sX[ROWS][LDSW];        // 135,168 B
  __shared__ bf16 sW[2][16*NB][LDSW];    // 135,168 B (double-buffered 64 N-rows)

  const int tid  = threadIdx.x;
  const int lane = tid & 31;
  const int l15  = lane & 15;
  const int half = lane >> 4;
  const int wrow = (tid >> 5) * 16;              // this wave's local M base
  const long m_base = (long)blockIdx.x * ROWS;

#ifdef USE_TDM
  // Prologue: start DMA of W phase 0 while we convert the X tile.
  if (tid < 32)
    tdm_load_w(Wsum, (unsigned)(unsigned long long)(void*)&sW[0][0][0]);
#endif

  // ---- stage X tile, fp32 -> bf16 (one conversion, reused 32x) ----
  {
    const float4* Xg = (const float4*)(X + m_base * H_);
    for (int idx = tid; idx < ROWS * (H_/4); idx += 256) {
      int r = idx >> 7;                 // H_/4 = 128 float4 per row
      int c = (idx & 127) << 2;
      float4 v = Xg[idx];
      v4bf p; p.x = (bf16)v.x; p.y = (bf16)v.y; p.z = (bf16)v.z; p.w = (bf16)v.w;
      *(v4bf*)&sX[r][c] = p;            // 8B DS store, aligned
    }
  }
  __syncthreads();                      // publish sX

  float part[8] = {0.f,0.f,0.f,0.f,0.f,0.f,0.f,0.f};
  const float vb = VaB[0];
  const int rowA = wrow + l15;

  for (int g = 0; g < GROUPS; ++g) {
    const int cur = g & 1;
    const int n0  = g * (16*NB);

#ifdef USE_TDM
    if (tid < 32) __builtin_amdgcn_s_wait_tensorcnt(0);   // TDM(g) landed
    __syncthreads();       // publish sW[cur]; all done reading sW[cur^1]
    if ((g + 1 < GROUPS) && (tid < 32))                   // overlap next DMA
      tdm_load_w(Wsum + (size_t)(n0 + 16*NB) * H_,
                 (unsigned)(unsigned long long)(void*)&sW[cur ^ 1][0][0]);
#else
    __syncthreads();                                      // done reading sW[cur]
    {
      const uint2* Wg = (const uint2*)(Wsum + (size_t)n0 * H_);
      for (int idx = tid; idx < 16*NB * (H_/4); idx += 256) {
        int r = idx >> 7;
        int c = (idx & 127) << 2;
        *(uint2*)&sW[cur][r][c] = Wg[idx];
      }
    }
    __syncthreads();
#endif

    float vaL[NB], bsL[NB];
#pragma unroll
    for (int j = 0; j < NB; ++j) {
      vaL[j] = VaW[n0 + j*16 + l15];
      bsL[j] = bsum[n0 + j*16 + l15];
    }

    v8f acc[NB];
#pragma unroll
    for (int j = 0; j < NB; ++j) acc[j] = (v8f){0.f,0.f,0.f,0.f,0.f,0.f,0.f,0.f};

#pragma unroll
    for (int k = 0; k < H_; k += 32) {
      // A fragment (16x32 bf16): lane<16 holds K [k,k+7]+[k+16,k+23]; lane>=16 +8
      const int klo = k + half * 8;
      v8bf a0 = *(const v8bf*)&sX[rowA][klo];
      v8bf a1 = *(const v8bf*)&sX[rowA][klo + 16];
      v16bf A = __builtin_shufflevector(a0, a1, 0,1,2,3,4,5,6,7,8,9,10,11,12,13,14,15);
      const int kb = k + half * 8;
#pragma unroll
      for (int j = 0; j < NB; ++j) {             // 4 WMMAs share one A fragment
        v8bf b0 = *(const v8bf*)&sW[cur][j*16 + l15][kb];
        v8bf b1 = *(const v8bf*)&sW[cur][j*16 + l15][kb + 16];
        v16bf Bm = __builtin_shufflevector(b0, b1, 0,1,2,3,4,5,6,7,8,9,10,11,12,13,14,15);
        acc[j] = __builtin_amdgcn_wmma_f32_16x16x32_bf16(false, A, false, Bm,
                                                         (short)0, acc[j], false, false);
      }
    }

    // epilogue: C VGPR i -> (M = wrow + half*8 + i, N = n0 + j*16 + l15)
#pragma unroll
    for (int j = 0; j < NB; ++j)
#pragma unroll
      for (int i = 0; i < 8; ++i) {
        float t = TANHF(acc[j][i] + bsL[j]);
        part[i] += t * vaL[j];
      }
  }

  // reduce over N: sum the 16 lanes of each half-wave
#pragma unroll
  for (int i = 0; i < 8; ++i) {
    float v = part[i];
    v += __shfl_xor(v, 1, 16);
    v += __shfl_xor(v, 2, 16);
    v += __shfl_xor(v, 4, 16);
    v += __shfl_xor(v, 8, 16);
    part[i] = v;
  }
  if (l15 == 0) {
    long rbase = m_base + wrow + half * 8;
#pragma unroll
    for (int i = 0; i < 8; ++i) score[rbase + i] = part[i] + vb;
  }
}

// ---------------- softmax over T per batch ---------------------------------
__global__ __launch_bounds__(256)
void softmax_kernel(const float* __restrict__ score, float* __restrict__ attn) {
  __shared__ float red[256];
  const int b = blockIdx.x, tid = threadIdx.x;
  const float* s = score + (size_t)b * T_;
  float* a = attn + (size_t)b * T_;

  float m = -1e30f;
  for (int j = tid; j < T_; j += 256) m = fmaxf(m, s[j]);
  red[tid] = m; __syncthreads();
  for (int off = 128; off > 0; off >>= 1) {
    if (tid < off) red[tid] = fmaxf(red[tid], red[tid + off]);
    __syncthreads();
  }
  m = red[0]; __syncthreads();

  float sum = 0.f;
  for (int j = tid; j < T_; j += 256) { float e = __expf(s[j] - m); a[j] = e; sum += e; }
  red[tid] = sum; __syncthreads();
  for (int off = 128; off > 0; off >>= 1) {
    if (tid < off) red[tid] += red[tid + off];
    __syncthreads();
  }
  const float inv = 1.0f / red[0];
  for (int j = tid; j < T_; j += 256) a[j] *= inv;
}

// ---------------- context = attn^T . X  ------------------------------------
__global__ __launch_bounds__(256)
void context_kernel(const float* __restrict__ X, const float* __restrict__ attn,
                    float* __restrict__ ctx) {
  const int b = blockIdx.x;
  const int h = blockIdx.y * 256 + threadIdx.x;
  const int t0 = blockIdx.z * (T_/8);
  const float* a = attn + (size_t)b * T_;
  float acc = 0.f;
  for (int t = t0; t < t0 + T_/8; ++t)
    acc += a[t] * X[((size_t)b * T_ + t) * H_ + h];
  atomicAdd(&ctx[(size_t)b * H_ + h], acc);
}

extern "C" void kernel_launch(void* const* d_in, const int* in_sizes, int n_in,
                              void* d_out, int out_size, void* d_ws, size_t ws_size,
                              hipStream_t stream) {
  const float* X   = (const float*)d_in[0];   // lstm_output [B,T,H]
  const float* WaW = (const float*)d_in[1];
  const float* WaB = (const float*)d_in[2];
  const float* UaW = (const float*)d_in[3];
  const float* UaB = (const float*)d_in[4];
  const float* VaW = (const float*)d_in[5];   // [1,H] -> H floats
  const float* VaB = (const float*)d_in[6];   // [1]

  float* ctx  = (float*)d_out;                       // [B,H]
  float* attn = (float*)d_out + (size_t)B_ * H_;     // [B,T]

  char*  ws    = (char*)d_ws;
  bf16*  Wsum  = (bf16*)ws;                                   // 512 KB
  float* bsum  = (float*)(ws + (size_t)H_*H_*2);              //   2 KB
  float* score = (float*)(ws + (size_t)H_*H_*2 + 4096);       //   1 MB

  prep_kernel   <<<(H_*H_)/256, 256, 0, stream>>>(WaW, UaW, WaB, UaB, Wsum, bsum, ctx);
  score_kernel  <<<M_/ROWS,     256, 0, stream>>>(X, Wsum, bsum, VaW, VaB, score);
  softmax_kernel<<<B_,          256, 0, stream>>>(score, attn);
  context_kernel<<<dim3(B_, H_/256, 8), 256, 0, stream>>>(X, attn, ctx);
}
